// VisionTransformerWSI_256_77043123356285
// MI455X (gfx1250) — compile-verified, matrix operand order checked by reading, not run on hardware
//
#include <hip/hip_runtime.h>
#include <hip/hip_bf16.h>

#define B_     8
#define NHEADS 12
#define HH     32
#define WW     32
#define DIM    768
#define HDIM   64
#define NTOK   1025
#define NPAD   1056            // 33 * 32, covers 1025 with WMMA-friendly padding
#define NQT    (NPAD / 16)     // 66 query tiles of 16
#define NMT    (NPAD / 32)     // 33 GEMM row-tiles of 32
#define NKT    (NPAD / 32)     // 33 key tiles of 32

typedef __attribute__((ext_vector_type(16))) __bf16 v16bf;
typedef __attribute__((ext_vector_type(8)))  float  v8f;

union BV { v16bf v; unsigned int u[8]; unsigned short s[16]; };

__device__ __forceinline__ unsigned short f2bf(float f) {
    unsigned int u = __float_as_uint(f);
    u += 0x7FFFu + ((u >> 16) & 1u);          // round-to-nearest-even
    return (unsigned short)(u >> 16);
}
__device__ __forceinline__ float bf2f(unsigned short h) {
    return __uint_as_float(((unsigned int)h) << 16);
}
// K-pair base for 16-bit WMMA A/B operand VGPR v (pairs of consecutive K)
__device__ __forceinline__ int kpair(int vv, int half) {
    return (vv < 4) ? (half * 8 + 2 * vv) : (16 + half * 8 + 2 * (vv - 4));
}
__device__ __forceinline__ void load_op(const unsigned short* __restrict__ row, int half, BV& dst) {
    #pragma unroll
    for (int vv = 0; vv < 8; ++vv)
        dst.u[vv] = *(const unsigned int*)(row + kpair(vv, half));
}

// ---------------- conversion / padding kernels ----------------

__global__ void k_conv_x(const float* __restrict__ x, unsigned short* __restrict__ xb) {
    long i = (long)blockIdx.x * 256 + threadIdx.x;
    long total = (long)B_ * NPAD * DIM;
    if (i >= total) return;
    int d = (int)(i % DIM);
    long t = i / DIM;
    int n = (int)(t % NPAD);
    int b = (int)(t / NPAD);
    float v = 0.f;
    if (n < NTOK) v = x[((long)b * NTOK + n) * DIM + d];
    xb[i] = f2bf(v);
}

__global__ void k_conv_qw(const float* __restrict__ w, unsigned short* __restrict__ wt) {
    int i = blockIdx.x * 256 + threadIdx.x;   // (2304 * 768), wt[n][k]
    if (i >= 3 * DIM * DIM) return;
    int k = i % DIM, n = i / DIM;
    wt[i] = f2bf(w[(long)k * (3 * DIM) + n]);
}

__global__ void k_conv_pw(const float* __restrict__ w, unsigned short* __restrict__ wt) {
    int i = blockIdx.x * 256 + threadIdx.x;   // (768 * 768), wt[n][k]
    if (i >= DIM * DIM) return;
    int k = i % DIM, n = i / DIM;
    wt[i] = f2bf(w[(long)k * DIM + n]);
}

// --- 32x64 WMMA macro-tile (bf16, f32 accum), double-buffered, B reused -----

__device__ __forceinline__ void wmma_tile32x64(const unsigned short* __restrict__ A, int lda,
                                               const unsigned short* __restrict__ Bt, int ldb,
                                               int K, v8f acc[2][4], int lane) {
    const int m = lane & 15, half = lane >> 4;
    BV a[2][2];       // [buffer][m-subtile]
    BV bm[2][4];      // [buffer][n-subtile]
    // prologue: chunk 0 -> buffer 0
    #pragma unroll
    for (int mi = 0; mi < 2; ++mi)
        load_op(A + (long)(mi * 16 + m) * lda, half, a[0][mi]);
    #pragma unroll
    for (int t = 0; t < 4; ++t)
        load_op(Bt + (long)(t * 16 + m) * ldb, half, bm[0][t]);
    #pragma unroll 2
    for (int k0 = 0; k0 < K; k0 += 32) {
        const int cur = (k0 >> 5) & 1, nxt = cur ^ 1;
        const int k1 = k0 + 32;
        if (k1 < K) {   // prefetch next chunk while current WMMAs run
            #pragma unroll
            for (int mi = 0; mi < 2; ++mi)
                load_op(A + (long)(mi * 16 + m) * lda + k1, half, a[nxt][mi]);
            #pragma unroll
            for (int t = 0; t < 4; ++t)
                load_op(Bt + (long)(t * 16 + m) * ldb + k1, half, bm[nxt][t]);
        }
        #pragma unroll
        for (int t = 0; t < 4; ++t)          // B operand reused across both m-subtiles
            #pragma unroll
            for (int mi = 0; mi < 2; ++mi)
                acc[mi][t] = __builtin_amdgcn_wmma_f32_16x16x32_bf16(
                    false, a[cur][mi].v, false, bm[cur][t].v, (short)0, acc[mi][t], false, false);
    }
}

// ---------------- QKV GEMM: xb(B,NPAD,768) @ Wt -> Q,K row-major, V transposed ----

__global__ void __launch_bounds__(128) k_qkv_gemm(
    const unsigned short* __restrict__ xb, const unsigned short* __restrict__ wt,
    const float* __restrict__ bias,
    unsigned short* __restrict__ qbuf, unsigned short* __restrict__ kbuf,
    unsigned short* __restrict__ vtbuf) {
    int lane = threadIdx.x & 31, wave = threadIdx.x >> 5;
    int b = blockIdx.z, mt = blockIdx.y;
    int n0 = (blockIdx.x * 4 + wave) * 64;
    const unsigned short* A = xb + ((long)b * NPAD + mt * 32) * DIM;
    v8f z = {0.f, 0.f, 0.f, 0.f, 0.f, 0.f, 0.f, 0.f};
    v8f acc[2][4] = {{z, z, z, z}, {z, z, z, z}};
    wmma_tile32x64(A, DIM, wt + (long)n0 * DIM, DIM, DIM, acc, lane);
    int m = lane & 15, half = lane >> 4;
    #pragma unroll
    for (int t = 0; t < 4; ++t) {
        int n = n0 + t * 16 + m;
        int which = n / DIM, rem = n % DIM, hd = rem >> 6, ch = rem & 63;
        float bv = bias[n];
        long bh = (long)b * NHEADS + hd;
        #pragma unroll
        for (int mi = 0; mi < 2; ++mi)
            #pragma unroll
            for (int r = 0; r < 8; ++r) {
                int qr = mt * 32 + mi * 16 + r + 8 * half;
                unsigned short o = f2bf(acc[mi][t][r] + bv);
                if (which == 0)      qbuf[(bh * NPAD + qr) * HDIM + ch]  = o;
                else if (which == 1) kbuf[(bh * NPAD + qr) * HDIM + ch]  = o;
                else                 vtbuf[(bh * HDIM + ch) * NPAD + qr] = o;
            }
    }
}

// ---------------- decomposed relative-position bias -----------------------

__global__ void k_relpos(const unsigned short* __restrict__ qbuf,
                         const float* __restrict__ rph, const float* __restrict__ rpw,
                         float* __restrict__ relh, float* __restrict__ relw) {
    long i = (long)blockIdx.x * 256 + threadIdx.x;   // (B*NHEADS, 1024, 32)
    long total = (long)B_ * NHEADS * HH * WW * 32;
    if (i >= total) return;
    int k = (int)(i & 31);
    long t = i >> 5;
    int p = (int)(t & 1023);
    long bh = t >> 10;
    int y = p >> 5, xc = p & 31;
    const unsigned short* q = qbuf + (bh * NPAD + 1 + p) * HDIM;
    const float* rh = rph + (long)(y - k + 31) * HDIM;
    const float* rw = rpw + (long)(xc - k + 31) * HDIM;
    float sh = 0.f, sw = 0.f;
    for (int c = 0; c < HDIM; ++c) {
        float qq = bf2f(q[c]);
        sh += qq * rh[c];
        sw += qq * rw[c];
    }
    relh[i] = sh;
    relw[i] = sw;
}

// ---------------- flash attention (1 wave = 16 query rows) -----------------

__global__ void __launch_bounds__(32) k_flash(
    const unsigned short* __restrict__ qbuf, const unsigned short* __restrict__ kbuf,
    const unsigned short* __restrict__ vtbuf, const unsigned char* __restrict__ mask,
    const float* __restrict__ relh, const float* __restrict__ relw,
    unsigned short* __restrict__ aout) {
    __shared__ float          sS[16][32];
    __shared__ unsigned short sP[16][32];
    __shared__ float          sF[16];
    __shared__ float          sL[16];
    const int lane = threadIdx.x;
    const int qt = blockIdx.x, h = blockIdx.y, b = blockIdx.z;
    const long bh = (long)b * NHEADS + h;
    const int m = lane & 15, half = lane >> 4;
    const float scale = 0.125f;   // HEAD_DIM^-0.5

    // Q as WMMA A-operand, two K-chunks of 32 channels
    BV qa[2];
    {
        const unsigned short* qr_ = qbuf + (bh * NPAD + qt * 16 + m) * HDIM;
        #pragma unroll
        for (int c = 0; c < 2; ++c)
            load_op(qr_ + c * 32, half, qa[c]);
    }
    v8f z = {0.f, 0.f, 0.f, 0.f, 0.f, 0.f, 0.f, 0.f};
    v8f o[4] = {z, z, z, z};
    float mrun = -3.0e38f, lrun = 0.f;   // lanes < 16 meaningful
    const int qown = qt * 16 + ((lane < 16) ? lane : 0);
    float mq = (qown < NTOK) ? (mask[(long)b * NTOK + qown] ? 1.f : 0.f) : 0.f;

    for (int kt = 0; kt < NKT; ++kt) {
        // ---- issue ALL K-operand loads, then back-to-back WMMAs ----
        BV kbv[2][2];
        #pragma unroll
        for (int t = 0; t < 2; ++t) {
            const unsigned short* kr = kbuf + (bh * NPAD + kt * 32 + t * 16 + m) * HDIM;
            #pragma unroll
            for (int c = 0; c < 2; ++c)
                load_op(kr + c * 32, half, kbv[t][c]);
        }
        v8f sc[2] = {z, z};
        #pragma unroll
        for (int t = 0; t < 2; ++t)
            #pragma unroll
            for (int c = 0; c < 2; ++c)
                sc[t] = __builtin_amdgcn_wmma_f32_16x16x32_bf16(
                    false, qa[c].v, false, kbv[t][c].v, (short)0, sc[t], false, false);
        #pragma unroll
        for (int t = 0; t < 2; ++t)
            #pragma unroll
            for (int r = 0; r < 8; ++r)
                sS[r + 8 * half][t * 16 + m] = sc[t][r];

        // ---- prefetch V operands now; latency hidden behind softmax ----
        BV vb[4];
        #pragma unroll
        for (int t = 0; t < 4; ++t) {
            const unsigned short* vr = vtbuf + (bh * HDIM + t * 16 + m) * NPAD + kt * 32;
            load_op(vr, half, vb[t]);
        }
        __syncthreads();

        // ---- online softmax: lane m owns row m ----
        if (lane < 16) {
            int qr = qt * 16 + lane;
            float sv[32];
            float rowmax = -3.0e38f;
            for (int k = 0; k < 32; ++k) {
                int j = kt * 32 + k;
                float s = -3.0e38f;
                if (j < NTOK && qr < NTOK) {
                    s = sS[lane][k] * scale;
                    if (j > 0 && qr > 0) {
                        long pidx = (bh * 1024 + (qr - 1)) * 32;
                        s += relh[pidx + ((j - 1) >> 5)] + relw[pidx + ((j - 1) & 31)];
                    }
                    float mj = mask[(long)b * NTOK + j] ? 1.f : 0.f;
                    s -= 1.0e9f * (1.f - mq * mj);
                }
                sv[k] = s;
                rowmax = fmaxf(rowmax, s);
            }
            float mnew = fmaxf(mrun, rowmax);
            float f, psum = 0.f;
            if (mnew <= -1.0e38f) {               // no valid key yet (pure pad rows)
                f = 1.f;
                for (int k = 0; k < 32; ++k) sP[lane][k] = 0;
            } else {
                f = __expf(mrun - mnew);
                for (int k = 0; k < 32; ++k) {
                    float p = (sv[k] <= -1.0e38f) ? 0.f : __expf(sv[k] - mnew);
                    psum += p;
                    sP[lane][k] = f2bf(p);
                }
            }
            mrun = mnew;
            lrun = lrun * f + psum;
            sF[lane] = f;
        }
        __syncthreads();

        // ---- O = O*f + P * V (V operands already resident) ----
        BV pa;
        #pragma unroll
        for (int vv = 0; vv < 8; ++vv)
            pa.u[vv] = *(const unsigned int*)(&sP[m][kpair(vv, half)]);
        #pragma unroll
        for (int t = 0; t < 4; ++t) {
            v8f c = o[t];
            #pragma unroll
            for (int r = 0; r < 8; ++r) c[r] *= sF[r + 8 * half];
            o[t] = __builtin_amdgcn_wmma_f32_16x16x32_bf16(
                false, pa.v, false, vb[t].v, (short)0, c, false, false);
        }
        __syncthreads();   // guard sS/sP/sF reuse next tile
    }

    if (lane < 16) sL[lane] = lrun;
    __syncthreads();
    #pragma unroll
    for (int t = 0; t < 4; ++t) {
        int ch = t * 16 + m;
        #pragma unroll
        for (int r = 0; r < 8; ++r) {
            int row = r + 8 * half, qr = qt * 16 + row;
            float val = 0.f;
            if (qr < NTOK) val = o[t][r] / sL[row];
            aout[((long)b * NPAD + qr) * DIM + h * HDIM + ch] = f2bf(val);
        }
    }
}

// ---------------- output projection GEMM -----------------------------------

__global__ void __launch_bounds__(128) k_proj_gemm(
    const unsigned short* __restrict__ aout, const unsigned short* __restrict__ wt,
    const float* __restrict__ bias, float* __restrict__ pout) {
    int lane = threadIdx.x & 31, wave = threadIdx.x >> 5;
    int b = blockIdx.z, mt = blockIdx.y;
    int n0 = (blockIdx.x * 4 + wave) * 64;
    const unsigned short* A = aout + ((long)b * NPAD + mt * 32) * DIM;
    v8f z = {0.f, 0.f, 0.f, 0.f, 0.f, 0.f, 0.f, 0.f};
    v8f acc[2][4] = {{z, z, z, z}, {z, z, z, z}};
    wmma_tile32x64(A, DIM, wt + (long)n0 * DIM, DIM, DIM, acc, lane);
    int m = lane & 15, half = lane >> 4;
    #pragma unroll
    for (int t = 0; t < 4; ++t) {
        int n = n0 + t * 16 + m;
        float bv = bias[n];
        #pragma unroll
        for (int mi = 0; mi < 2; ++mi)
            #pragma unroll
            for (int r = 0; r < 8; ++r) {
                int qr = mt * 32 + mi * 16 + r + 8 * half;
                if (qr < NTOK)
                    pout[((long)b * NPAD + qr) * DIM + n] = acc[mi][t][r] + bv;
            }
    }
}

// ---------------- masked avg + max pooling ----------------------------------

__global__ void k_pool(const float* __restrict__ pout, const unsigned char* __restrict__ mask,
                       float* __restrict__ out) {
    int i = blockIdx.x * 256 + threadIdx.x;   // B * DIM
    if (i >= B_ * DIM) return;
    int d = i % DIM, b = i / DIM;
    float s = 0.f, cnt = 0.f, mx = -3.4e38f;
    for (int n = 1; n < NTOK; ++n) {
        float v = pout[((long)b * NPAD + n) * DIM + d];
        float mm = mask[(long)b * NTOK + n] ? 1.f : 0.f;
        s += v * mm;
        cnt += mm;
        mx = fmaxf(mx, v);
    }
    out[(long)b * 2 * DIM + d] = s / (cnt + 1e-7f);
    out[(long)b * 2 * DIM + DIM + d] = mx;
}

// ---------------- host launcher ---------------------------------------------

extern "C" void kernel_launch(void* const* d_in, const int* in_sizes, int n_in,
                              void* d_out, int out_size, void* d_ws, size_t ws_size,
                              hipStream_t stream) {
    const float*         x      = (const float*)d_in[0];
    const unsigned char* mask   = (const unsigned char*)d_in[1];
    const float*         qkv_w  = (const float*)d_in[2];
    const float*         qkv_b  = (const float*)d_in[3];
    const float*         proj_w = (const float*)d_in[4];
    const float*         proj_b = (const float*)d_in[5];
    const float*         rph    = (const float*)d_in[6];
    const float*         rpw    = (const float*)d_in[7];
    float* out = (float*)d_out;

    char* ws = (char*)d_ws;
    size_t off = 0;
    auto alloc = [&](size_t bytes) -> void* {
        void* p = ws + off;
        off += (bytes + 255) & ~(size_t)255;
        return p;
    };
    unsigned short* xb     = (unsigned short*)alloc((size_t)B_ * NPAD * DIM * 2);
    unsigned short* qkvwt  = (unsigned short*)alloc((size_t)3 * DIM * DIM * 2);
    unsigned short* projwt = (unsigned short*)alloc((size_t)DIM * DIM * 2);
    unsigned short* qbuf   = (unsigned short*)alloc((size_t)B_ * NHEADS * NPAD * HDIM * 2);
    unsigned short* kbuf   = (unsigned short*)alloc((size_t)B_ * NHEADS * NPAD * HDIM * 2);
    unsigned short* vtbuf  = (unsigned short*)alloc((size_t)B_ * NHEADS * NPAD * HDIM * 2);
    float*          relh   = (float*)alloc((size_t)B_ * NHEADS * 1024 * 32 * 4);
    float*          relw   = (float*)alloc((size_t)B_ * NHEADS * 1024 * 32 * 4);
    unsigned short* aout   = (unsigned short*)alloc((size_t)B_ * NPAD * DIM * 2);
    float*          pout   = (float*)alloc((size_t)B_ * NPAD * DIM * 4);

    // 1) convert / pad / transpose inputs to bf16
    {
        long n = (long)B_ * NPAD * DIM;
        k_conv_x<<<dim3((unsigned)((n + 255) / 256)), dim3(256), 0, stream>>>(x, xb);
    }
    k_conv_qw<<<dim3((3 * DIM * DIM + 255) / 256), dim3(256), 0, stream>>>(qkv_w, qkvwt);
    k_conv_pw<<<dim3((DIM * DIM + 255) / 256), dim3(256), 0, stream>>>(proj_w, projwt);

    // 2) QKV GEMM (bf16 WMMA): 2304 cols = 9 blocks * 4 waves * 64; 33 row tiles of 32
    k_qkv_gemm<<<dim3(9, NMT, B_), dim3(128), 0, stream>>>(xb, qkvwt, qkv_b, qbuf, kbuf, vtbuf);

    // 3) relative-position bias components
    {
        long n = (long)B_ * NHEADS * HH * WW * 32;
        k_relpos<<<dim3((unsigned)((n + 255) / 256)), dim3(256), 0, stream>>>(qbuf, rph, rpw, relh, relw);
    }

    // 4) flash attention (WMMA QK^T and PV)
    k_flash<<<dim3(NQT, NHEADS, B_), dim3(32), 0, stream>>>(qbuf, kbuf, vtbuf, mask, relh, relw, aout);

    // 5) output projection GEMM: 768 cols = 3 blocks * 4 waves * 64; 33 row tiles of 32
    k_proj_gemm<<<dim3(3, NMT, B_), dim3(128), 0, stream>>>(aout, projwt, proj_b, pout);

    // 6) masked avg + max pooling
    k_pool<<<dim3((B_ * DIM + 255) / 256), dim3(256), 0, stream>>>(pout, mask, out);
}
